// CharRNN_88459146428802
// MI455X (gfx1250) — matrix operand order repaired and checked by reading, not compiled, not run on hardware
//
#include <hip/hip_runtime.h>
#include <hip/hip_bf16.h>
#include <math.h>

// ---------------- problem constants ----------------
#define SEQ_    1024
#define BATCH_  128
#define HIDDEN_ 2048
#define EMBED_  512
#define VOCAB_  50257
#define EULER_  0.01f
#define DIFF_   0.001f

#define NBLK_SCAN 256      // persistent blocks; 256 blk * 4 waves = 1024 tiles = (128/16)*(2048/16)

typedef float  v2f   __attribute__((ext_vector_type(2)));
typedef float  v8f   __attribute__((ext_vector_type(8)));
typedef __bf16 v8bf  __attribute__((ext_vector_type(8)));
typedef __bf16 v16bf __attribute__((ext_vector_type(16)));

__device__ __forceinline__ v16bf cat16(v8bf a, v8bf b) {
    return __builtin_shufflevector(a, b, 0,1,2,3,4,5,6,7,8,9,10,11,12,13,14,15);
}

// ---------------------------------------------------------------------------
// Prep: split W_eff^T into bf16 hi/lo planes in a WMMA-B swizzle (contiguous k = 16h + s order):
//   Wb[((k>>4)*H + col)*16 + (k&15)] = bf16 of W_eff[col][k]
// so a lane's 16 consecutive K values for one column are contiguous (2x b128).
// ---------------------------------------------------------------------------
__global__ void prep_weff_kernel(const float* __restrict__ Wh,
                                 __bf16* __restrict__ Wb_hi, __bf16* __restrict__ Wb_lo) {
    int idx = blockIdx.x * 256 + threadIdx.x;           // idx = k*H + col
    if (idx >= HIDDEN_ * HIDDEN_) return;
    int k = idx / HIDDEN_;
    int col = idx - k * HIDDEN_;
    float w = Wh[col * HIDDEN_ + k] - Wh[k * HIDDEN_ + col];
    if (k == col) w -= DIFF_;
    size_t pos = ((size_t)(k >> 4) * HIDDEN_ + col) * 16 + (k & 15);
    __bf16 hi = (__bf16)w;
    Wb_hi[pos] = hi;
    Wb_lo[pos] = (__bf16)(w - (float)hi);
}

// Same swizzled hi/lo split for W_in^T (K = EMBED rows, H cols).
__global__ void prep_wint_kernel(const float* __restrict__ Win,
                                 __bf16* __restrict__ Wb_hi, __bf16* __restrict__ Wb_lo) {
    int idx = blockIdx.x * 256 + threadIdx.x;           // idx = k*H + col
    if (idx >= EMBED_ * HIDDEN_) return;
    int k = idx / HIDDEN_;
    int col = idx - k * HIDDEN_;
    float w = Win[col * EMBED_ + k];
    size_t pos = ((size_t)(k >> 4) * HIDDEN_ + col) * 16 + (k & 15);
    __bf16 hi = (__bf16)w;
    Wb_hi[pos] = hi;
    Wb_lo[pos] = (__bf16)(w - (float)hi);
}

// Seed fp32 carry buffer + bf16 hi/lo planes from h_init.
__global__ void prep_h0_kernel(const float* __restrict__ h0, float* __restrict__ hB,
                               __bf16* __restrict__ phi, __bf16* __restrict__ plo) {
    int idx = blockIdx.x * 256 + threadIdx.x;
    if (idx >= BATCH_ * HIDDEN_) return;
    float v = h0[idx];
    hB[idx] = v;
    __bf16 hi = (__bf16)v;
    phi[idx] = hi;
    plo[idx] = (__bf16)(v - (float)hi);
}

// ---------------------------------------------------------------------------
// Persistent scan: 1024 Euler steps, device-wide barrier between steps.
// Each wave owns one 16x16 tile of h. Split-precision bf16x2:
//   C += Ahi*Bhi + Ahi*Blo + Alo*Bhi   via V_WMMA_F32_16X16X32_BF16
// fp32 carry of h; bf16 planes re-emitted each step for the next A operand.
// ---------------------------------------------------------------------------
__global__ void __launch_bounds__(128, 1)
scan_kernel(float* __restrict__ hA, float* __restrict__ hB,
            __bf16* __restrict__ pAhi, __bf16* __restrict__ pAlo,
            __bf16* __restrict__ pBhi, __bf16* __restrict__ pBlo,
            const __bf16* __restrict__ Wb_hi, const __bf16* __restrict__ Wb_lo,
            const __bf16* __restrict__ Wib_hi, const __bf16* __restrict__ Wib_lo,
            const float* __restrict__ bias, const float* __restrict__ emb,
            const int*   __restrict__ ids, unsigned* __restrict__ bar)
{
    const int lane  = threadIdx.x & 31;
    const int wave  = threadIdx.x >> 5;
    const int tile  = blockIdx.x * 4 + wave;     // 0..1023
    const int row0  = (tile >> 7) * 16;          // batch rows
    const int col0  = (tile & 127) * 16;         // hidden cols
    const int m     = lane & 15;
    const int khalf = lane >> 4;                 // 0/1: which 16-K half this lane holds
    const int koff  = khalf * 16;                // A-operand K sub-offset
    const int rowadd = khalf * 8;                // C/D row offset for high lanes
    const float bias_n = bias[col0 + m];

    // B-operand bases (swizzled): lane column = col0+m, lane k-block offset = khalf
    const __bf16* bWh = Wb_hi  + ((size_t)khalf * HIDDEN_ + col0 + m) * 16;
    const __bf16* bWl = Wb_lo  + ((size_t)khalf * HIDDEN_ + col0 + m) * 16;
    const __bf16* bIh = Wib_hi + ((size_t)khalf * HIDDEN_ + col0 + m) * 16;
    const __bf16* bIl = Wib_lo + ((size_t)khalf * HIDDEN_ + col0 + m) * 16;

    for (int s = 0; s < SEQ_; ++s) {
        const float*  hin   = (s & 1) ? hA : hB;
        float*        hout  = (s & 1) ? hB : hA;
        const __bf16* pih   = (s & 1) ? pAhi : pBhi;
        const __bf16* pil   = (s & 1) ? pAlo : pBlo;
        __bf16*       poh   = (s & 1) ? pBhi : pAhi;
        __bf16*       pol   = (s & 1) ? pBlo : pAlo;

        v8f c = {0.f, 0.f, 0.f, 0.f, 0.f, 0.f, 0.f, 0.f};

        // ---- recurrent GEMM: K = HIDDEN, step 32 ----
        {
            const __bf16* ah = pih + (row0 + m) * HIDDEN_ + koff;
            const __bf16* al = pil + (row0 + m) * HIDDEN_ + koff;
#pragma unroll 2
            for (int k = 0; k < HIDDEN_; k += 32) {
                const v8bf* pa = (const v8bf*)(ah + k);
                const v8bf* pl = (const v8bf*)(al + k);
                v16bf ahi = cat16(pa[0], pa[1]);       // K: k+koff..+15
                v16bf alo = cat16(pl[0], pl[1]);
                const v8bf* qh = (const v8bf*)(bWh + (size_t)(k >> 4) * (HIDDEN_ * 16));
                const v8bf* ql = (const v8bf*)(bWl + (size_t)(k >> 4) * (HIDDEN_ * 16));
                v16bf bhi = cat16(qh[0], qh[1]);
                v16bf blo = cat16(ql[0], ql[1]);
                c = __builtin_amdgcn_wmma_f32_16x16x32_bf16(false, ahi, false, bhi, (short)0, c, false, false);
                c = __builtin_amdgcn_wmma_f32_16x16x32_bf16(false, ahi, false, blo, (short)0, c, false, false);
                c = __builtin_amdgcn_wmma_f32_16x16x32_bf16(false, alo, false, bhi, (short)0, c, false, false);
            }
        }

        // ---- fused input projection: u = emb[ids[s]] @ W_in^T, K = EMBED ----
        {
            const int id = ids[s * BATCH_ + row0 + m];
            const float* ep = emb + (size_t)id * EMBED_ + koff;
#pragma unroll 2
            for (int k = 0; k < EMBED_; k += 32) {
                v16bf ahi, alo;
#pragma unroll
                for (int j = 0; j < 8; ++j) {          // split gathered fp32 on the fly
                    float f0 = ep[k + j];
                    float f1 = ep[k + 8 + j];
                    __bf16 h0b = (__bf16)f0;
                    __bf16 h1b = (__bf16)f1;
                    ahi[j]     = h0b; alo[j]     = (__bf16)(f0 - (float)h0b);
                    ahi[8 + j] = h1b; alo[8 + j] = (__bf16)(f1 - (float)h1b);
                }
                const v8bf* qh = (const v8bf*)(bIh + (size_t)(k >> 4) * (HIDDEN_ * 16));
                const v8bf* ql = (const v8bf*)(bIl + (size_t)(k >> 4) * (HIDDEN_ * 16));
                v16bf bhi = cat16(qh[0], qh[1]);
                v16bf blo = cat16(ql[0], ql[1]);
                c = __builtin_amdgcn_wmma_f32_16x16x32_bf16(false, ahi, false, bhi, (short)0, c, false, false);
                c = __builtin_amdgcn_wmma_f32_16x16x32_bf16(false, ahi, false, blo, (short)0, c, false, false);
                c = __builtin_amdgcn_wmma_f32_16x16x32_bf16(false, alo, false, bhi, (short)0, c, false, false);
            }
        }

        // ---- Euler update (fp32 carry) + emit bf16 planes for next step ----
#pragma unroll
        for (int r = 0; r < 8; ++r) {
            const int row = row0 + rowadd + r;
            const int idx = row * HIDDEN_ + col0 + m;
            float hv = hin[idx] + EULER_ * (c[r] + bias_n);
            hout[idx] = hv;
            __bf16 hb = (__bf16)hv;
            poh[idx] = hb;
            pol[idx] = (__bf16)(hv - (float)hb);
        }

        // ---- device-wide barrier (monotone counter) ----
        __threadfence();
        __syncthreads();
        if (threadIdx.x == 0) {
            __hip_atomic_fetch_add(bar, 1u, __ATOMIC_RELEASE, __HIP_MEMORY_SCOPE_AGENT);
            const unsigned target = (unsigned)(s + 1) * NBLK_SCAN;
            while (__hip_atomic_load(bar, __ATOMIC_ACQUIRE, __HIP_MEMORY_SCOPE_AGENT) < target)
                __builtin_amdgcn_s_sleep(2);
        }
        __syncthreads();
        __threadfence();
    }
}

// ---------------------------------------------------------------------------
// Logits: [128 x 2048] @ fc_w^T + fc_b, /temperature -> d_out (fp32 WMMA;
// memory-bound on fc_w's 412 MB stream, lane-contiguous b64 B loads).
// ---------------------------------------------------------------------------
__global__ void __launch_bounds__(256)
logits_kernel(const float* __restrict__ h, const float* __restrict__ fcw,
              const float* __restrict__ fcb, const int* __restrict__ tptr,
              float* __restrict__ out)
{
    const int NT = (VOCAB_ + 15) / 16;
    const int tile = blockIdx.x * 8 + (threadIdx.x >> 5);
    if (tile >= 8 * NT) return;
    const int bt = tile / NT;
    const int vt = tile - bt * NT;
    const int row0 = bt * 16;
    const int col0 = vt * 16;
    const int lane = threadIdx.x & 31;
    const int m    = lane & 15;
    const int koff = (lane >> 4) * 2;
    const int rowadd = (lane >> 4) * 8;

    const int v  = col0 + m;
    const int vc = (v < VOCAB_) ? v : (VOCAB_ - 1);

    v8f c = {0.f, 0.f, 0.f, 0.f, 0.f, 0.f, 0.f, 0.f};
    const float* aptr = h + (row0 + m) * HIDDEN_ + koff;
    const float* bptr = fcw + (size_t)vc * HIDDEN_ + koff;
#pragma unroll 4
    for (int k = 0; k < HIDDEN_; k += 4) {
        v2f a; a.x = aptr[k]; a.y = aptr[k + 1];
        v2f b; b.x = bptr[k]; b.y = bptr[k + 1];
        c = __builtin_amdgcn_wmma_f32_16x16x4_f32(false, a, false, b, (short)0, c, false, false);
    }

    const float invT = 1.0f / (float)(*tptr);
    if (v < VOCAB_) {
        const float bv = fcb[v];
#pragma unroll
        for (int r = 0; r < 8; ++r) {
            const int row = row0 + rowadd + r;
            out[(size_t)row * VOCAB_ + v] = (c[r] + bv) * invT;
        }
    }
}

// ---------------------------------------------------------------------------
__global__ void __launch_bounds__(256)
softmax_kernel(float* __restrict__ out) {
    const int b = blockIdx.x;
    float* row = out + (size_t)b * VOCAB_;
    __shared__ float red[256];
    const int tid = threadIdx.x;

    float lmax = -INFINITY;
    for (int v = tid; v < VOCAB_; v += 256) lmax = fmaxf(lmax, row[v]);
    red[tid] = lmax; __syncthreads();
    for (int s = 128; s > 0; s >>= 1) {
        if (tid < s) red[tid] = fmaxf(red[tid], red[tid + s]);
        __syncthreads();
    }
    const float mx = red[0]; __syncthreads();

    float lsum = 0.f;
    for (int v = tid; v < VOCAB_; v += 256) lsum += expf(row[v] - mx);
    red[tid] = lsum; __syncthreads();
    for (int s = 128; s > 0; s >>= 1) {
        if (tid < s) red[tid] += red[tid + s];
        __syncthreads();
    }
    const float inv = 1.0f / red[0]; __syncthreads();

    for (int v = tid; v < VOCAB_; v += 256) row[v] = expf(row[v] - mx) * inv;
}

// ---------------------------------------------------------------------------
extern "C" void kernel_launch(void* const* d_in, const int* in_sizes, int n_in,
                              void* d_out, int out_size, void* d_ws, size_t ws_size,
                              hipStream_t stream) {
    const int*   ids  = (const int*)  d_in[0];
    const float* h0   = (const float*)d_in[1];
    const float* emb  = (const float*)d_in[2];
    const float* Win  = (const float*)d_in[3];
    const float* Wh   = (const float*)d_in[4];
    const float* bias = (const float*)d_in[5];
    const float* fcw  = (const float*)d_in[6];
    const float* fcb  = (const float*)d_in[7];
    const int*   temp = (const int*)  d_in[8];
    float* out = (float*)d_out;

    // carve workspace (256B aligned chunks)
    char* w = (char*)d_ws;
    auto alloc = [&](size_t bytes) { char* p = w; w += (bytes + 255) & ~(size_t)255; return p; };
    __bf16* Wb_hi  = (__bf16*)alloc((size_t)HIDDEN_ * HIDDEN_ * 2);
    __bf16* Wb_lo  = (__bf16*)alloc((size_t)HIDDEN_ * HIDDEN_ * 2);
    __bf16* Wib_hi = (__bf16*)alloc((size_t)EMBED_  * HIDDEN_ * 2);
    __bf16* Wib_lo = (__bf16*)alloc((size_t)EMBED_  * HIDDEN_ * 2);
    float*  hA     = (float*) alloc((size_t)BATCH_  * HIDDEN_ * 4);
    float*  hB     = (float*) alloc((size_t)BATCH_  * HIDDEN_ * 4);
    __bf16* pAhi   = (__bf16*)alloc((size_t)BATCH_  * HIDDEN_ * 2);
    __bf16* pAlo   = (__bf16*)alloc((size_t)BATCH_  * HIDDEN_ * 2);
    __bf16* pBhi   = (__bf16*)alloc((size_t)BATCH_  * HIDDEN_ * 2);
    __bf16* pBlo   = (__bf16*)alloc((size_t)BATCH_  * HIDDEN_ * 2);
    unsigned* bar  = (unsigned*)alloc(256);

    hipMemsetAsync(bar, 0, 64, stream);

    prep_weff_kernel<<<(HIDDEN_ * HIDDEN_ + 255) / 256, 256, 0, stream>>>(Wh, Wb_hi, Wb_lo);
    prep_wint_kernel<<<(EMBED_ * HIDDEN_ + 255) / 256, 256, 0, stream>>>(Win, Wib_hi, Wib_lo);
    prep_h0_kernel<<<(BATCH_ * HIDDEN_ + 255) / 256, 256, 0, stream>>>(h0, hB, pBhi, pBlo);

    scan_kernel<<<NBLK_SCAN, 128, 0, stream>>>(hA, hB, pAhi, pAlo, pBhi, pBlo,
                                               Wb_hi, Wb_lo, Wib_hi, Wib_lo,
                                               bias, emb, ids, bar);

    // SEQ_ = 1024 (even) -> final fp32 state is in hB
    const int NT = (VOCAB_ + 15) / 16;
    logits_kernel<<<NT, 256, 0, stream>>>(hB, fcw, fcb, temp, out);
    softmax_kernel<<<BATCH_, 256, 0, stream>>>(out);
}
